// HyperbolicLorentzDynamics_23871428232051
// MI455X (gfx1250) — compile-verified, matrix operand order checked by reading, not run on hardware
//
#include <hip/hip_runtime.h>
#include <hip/hip_bf16.h>

typedef __attribute__((ext_vector_type(16))) _Float16 v16h;
typedef __attribute__((ext_vector_type(8)))  float    v8f;

#define DP1 129          // D+1
#define ROWS_PER_BLOCK 128
#define NWAVES 8
// LDS layout (bytes):
//   [0,      32768) : W_eff, fragment-major f16 (32 frags x 512 halves)
//   [32768,  67584) : per-wave A tiles, 16 rows x 136 halves (pad 8) = 4352 B each
//   [67584, 135168) : per-wave velocity tiles, 16 rows x 132 floats (pad 4) = 8448 B each
#define SMEM_BYTES 135168
#define A_TILE_OFF 32768
#define V_TILE_OFF 67584
#define A_STRIDE 136
#define V_STRIDE 132

// ---------------------------------------------------------------------------
// Kernel 1: collapse W_temp(128x96) @ W_lt(96x128) -> W_eff(128x128) in f16,
// stored directly in WMMA B-fragment order: frag(kt,nt) -> 32 lanes x 16 halves.
// Half j of lane l in fragment (kt,nt) holds W_eff[K][N] with
//   K = kt*32 + (j&8 ? 16:0) + (j&7) + (l>=16 ? 8:0),  N = nt*16 + (l&15)
// (mirrors the documented 16-bit A/B VGPR striping for V_WMMA_*_16X16X32).
// Also b_eff[n] = sum_s b_temp[s]*W_lt[s][n] + b_lt[n].
// ---------------------------------------------------------------------------
__global__ __launch_bounds__(256)
void hld_prep_weights(const float* __restrict__ W_temp, const float* __restrict__ b_temp,
                      const float* __restrict__ W_lt,   const float* __restrict__ b_lt,
                      _Float16* __restrict__ weff, float* __restrict__ beff) {
  int tid  = blockIdx.x * 256 + threadIdx.x;    // 0..16383
  int frag = tid >> 9;
  int rem  = tid & 511;
  int lane = rem >> 4;
  int j    = rem & 15;
  int kt = frag >> 3, nt = frag & 7;
  int K = kt * 32 + ((j & 8) ? 16 : 0) + (j & 7) + ((lane >= 16) ? 8 : 0);
  int N = nt * 16 + (lane & 15);
  float acc = 0.f;
  #pragma unroll 8
  for (int s = 0; s < 96; ++s) acc += W_temp[K * 96 + s] * W_lt[s * 128 + N];
  weff[tid] = (_Float16)acc;

  if (blockIdx.x == 0 && threadIdx.x < 128) {
    int n = threadIdx.x;
    float b = b_lt[n];
    #pragma unroll 8
    for (int s = 0; s < 96; ++s) b += b_temp[s] * W_lt[s * 128 + n];
    beff[n] = b;
  }
}

// ---------------------------------------------------------------------------
// Kernel 2: fully fused Lorentz dynamics. 8 waves/block, 16 rows/wave,
// 2 lanes per row (lane = 16*h + r, h selects spatial elements 1..64 / 65..128).
// ---------------------------------------------------------------------------
__global__ __launch_bounds__(256)
void hld_main(const float* __restrict__ xc_g, const float* __restrict__ xp_g,
              const _Float16* __restrict__ weff, const float* __restrict__ beff,
              const float* __restrict__ alpha_p, const float* __restrict__ vscale_p,
              float* __restrict__ out_next, float* __restrict__ out_xc) {
  extern __shared__ char smem[];
  _Float16* weff_s = (_Float16*)smem;
  const int tid  = threadIdx.x;
  const int wave = tid >> 5;
  const int lane = tid & 31;
  const int r    = lane & 15;   // row in wave tile / B-frag column lane
  const int h    = lane >> 4;   // element half
  _Float16* atile = (_Float16*)(smem + A_TILE_OFF + wave * (16 * A_STRIDE * 2));
  float*    vtile = (float*)   (smem + V_TILE_OFF + wave * (16 * V_STRIDE * 4));

  // Stage W_eff fragments to LDS: 32 KB as 2048 uint4, coalesced.
  {
    const uint4* src = (const uint4*)weff;
    uint4*       dst = (uint4*)weff_s;
    #pragma unroll
    for (int i = 0; i < 8; ++i) dst[tid + i * 256] = src[tid + i * 256];
  }

  const float sig_v  = 1.f / (1.f + __expf(-vscale_p[0]));
  const float vmul   = 0.3f * sig_v;
  const float alpha  = 1.f / (1.f + __expf(-alpha_p[0]));
  const float om_a   = 1.f - alpha;

  const long  row0 = (long)blockIdx.x * ROWS_PER_BLOCK + wave * 16 + r;
  const float* xcr = xc_g + row0 * DP1;
  const float* xpr = xp_g + row0 * DP1;
  const int   i0   = 1 + h * 64;   // first ambient index this lane owns

  float xcv[64];   // this lane's 64 spatial components of x_current (live throughout)
  float work[64];  // x_prev -> velocity -> v_tan -> x_update -> tangent -> x_next
  float xc0 = xcr[0];
  float xc2;       // ||x_current spatial||^2 (row-wide)

  // ---- Phase 1: traj_sp = logmap(x_prev, x_cur)[1:], written as f16 A tile ----
  {
    float xp0 = xpr[0];
    float pip = 0.f, pxc2 = 0.f;
    #pragma unroll
    for (int c = 0; c < 64; ++c) { xcv[c] = xcr[i0 + c]; work[c] = xpr[i0 + c]; }
    #pragma unroll
    for (int c = 0; c < 64; ++c) { pip += xcv[c] * work[c]; pxc2 += xcv[c] * xcv[c]; }
    if (h == 0) pip -= xp0 * xc0;                 // linner: spatial sum - u0*v0
    float ip = pip + __shfl_xor(pip, 16, 32);
    xc2      = pxc2 + __shfl_xor(pxc2, 16, 32);
    float t  = fmaxf(-ip, 1.0f + 1e-7f);
    float d  = acoshf(t);
    // linner(u,u) = ip^2 - 1 for u = y + ip*x on the manifold
    float un = sqrtf(fmaxf(ip * ip - 1.f, 1e-8f));
    float sc = d / un;
    #pragma unroll
    for (int c = 0; c < 64; ++c)
      atile[r * A_STRIDE + h * 64 + c] = (_Float16)(sc * (xcv[c] + ip * work[c]));
  }
  __syncthreads();   // weff_s staged + A tiles visible

  // ---- Phase 2: velocity = clip(traj_sp @ W_eff + b_eff, +-5) * 0.3*sigmoid ----
  #pragma unroll
  for (int nt = 0; nt < 8; ++nt) {
    v8f acc = {0.f, 0.f, 0.f, 0.f, 0.f, 0.f, 0.f, 0.f};
    #pragma unroll
    for (int kt = 0; kt < 4; ++kt) {
      v16h a, b;
      #pragma unroll
      for (int j = 0; j < 16; ++j) {
        int K = kt * 32 + ((j & 8) ? 16 : 0) + (j & 7) + (h ? 8 : 0);
        a[j] = atile[r * A_STRIDE + K];
      }
      const _Float16* bf = weff_s + (kt * 8 + nt) * 512 + lane * 16;
      #pragma unroll
      for (int j = 0; j < 16; ++j) b[j] = bf[j];
      acc = __builtin_amdgcn_wmma_f32_16x16x32_f16(false, a, false, b,
                                                   (short)0, acc, false, false);
    }
    float be = beff[nt * 16 + (lane & 15)];
    #pragma unroll
    for (int v = 0; v < 8; ++v) {
      float val = fminf(fmaxf(acc[v] + be, -5.f), 5.f) * vmul;
      int row = v + (h ? 8 : 0);                       // D-matrix: VGPR v -> row
      vtile[row * V_STRIDE + nt * 16 + (lane & 15)] = val;
    }
  }
  __syncthreads();

  // ---- Phase 3: fused epilogue (register arrays + shfl reductions) ----
  #pragma unroll
  for (int c = 0; c < 64; ++c) work[c] = vtile[r * V_STRIDE + h * 64 + c];

  // s1 = linner(x_cur, v_amb) with v_amb[0]=0
  float p = 0.f;
  #pragma unroll
  for (int c = 0; c < 64; ++c) p += xcv[c] * work[c];
  float s1 = p + __shfl_xor(p, 16, 32);

  // v_tan = v_amb + s1*x_cur ; nn = linner(v_tan, v_tan) = sum_sp - vt0^2
  float vt0 = s1 * xc0;
  float pn = (h == 0) ? -vt0 * vt0 : 0.f;
  #pragma unroll
  for (int c = 0; c < 64; ++c) { work[c] += s1 * xcv[c]; pn += work[c] * work[c]; }
  float nn  = pn + __shfl_xor(pn, 16, 32);
  float un  = sqrtf(fmaxf(nn, 1e-8f));
  float unc = fminf(un, 10.f);
  float chv = coshf(unc), shv = sinhf(unc) / un;

  // x_update spatial = cosh*xc + sinh*vt/un ; then projx (time recomputed)
  float ps = 0.f;
  #pragma unroll
  for (int c = 0; c < 64; ++c) { work[c] = chv * xcv[c] + shv * work[c]; ps += work[c] * work[c]; }
  float S    = ps + __shfl_xor(ps, 16, 32);
  float xu0  = sqrtf(1.f + S);
  float xc0p = sqrtf(1.f + xc2);          // projx(x_current) time coord

  // tangent = logmap(xc, xu)
  p = 0.f;
  #pragma unroll
  for (int c = 0; c < 64; ++c) p += xcv[c] * work[c];
  float ip2 = p + __shfl_xor(p, 16, 32) - xc0p * xu0;
  float t2  = fmaxf(-ip2, 1.0f + 1e-7f);
  float d2  = acoshf(t2);
  float un2 = sqrtf(fmaxf(ip2 * ip2 - 1.f, 1e-8f));
  float f   = om_a * d2 / un2;            // (1-alpha)*d2/un2, scales u2

  // s = (1-alpha)*tangent ; ls = linner(s,s) = sum_sp - s0^2
  float s0 = f * (xu0 + ip2 * xc0p);
  float pl = (h == 0) ? -s0 * s0 : 0.f;
  #pragma unroll
  for (int c = 0; c < 64; ++c) { work[c] = f * (work[c] + ip2 * xcv[c]); pl += work[c] * work[c]; }
  float ls   = pl + __shfl_xor(pl, 16, 32);
  float un3  = sqrtf(fmaxf(ls, 1e-8f));
  float unc3 = fminf(un3, 10.f);
  float ch3  = coshf(unc3), sh3 = sinhf(unc3) / un3;

  // x_next spatial = cosh*xc + sinh*s/un3 ; projx time
  float pq = 0.f;
  #pragma unroll
  for (int c = 0; c < 64; ++c) { work[c] = ch3 * xcv[c] + sh3 * work[c]; pq += work[c] * work[c]; }
  float Q   = pq + __shfl_xor(pq, 16, 32);
  float xn0 = sqrtf(1.f + Q);

  float* on = out_next + row0 * DP1;
  float* oc = out_xc   + row0 * DP1;
  if (h == 0) { on[0] = xn0; oc[0] = xc0p; }
  #pragma unroll
  for (int c = 0; c < 64; ++c) { on[i0 + c] = work[c]; oc[i0 + c] = xcv[c]; }
}

extern "C" void kernel_launch(void* const* d_in, const int* in_sizes, int n_in,
                              void* d_out, int out_size, void* d_ws, size_t ws_size,
                              hipStream_t stream) {
  const float* xc     = (const float*)d_in[0];   // x_current  [B,129]
  const float* xp     = (const float*)d_in[1];   // x_previous [B,129]
  const float* W_temp = (const float*)d_in[2];   // [128,96]
  const float* b_temp = (const float*)d_in[3];   // [96]
  const float* W_lt   = (const float*)d_in[4];   // [96,128]
  const float* b_lt   = (const float*)d_in[5];   // [128]
  const float* alpha  = (const float*)d_in[6];   // scalar
  const float* vscale = (const float*)d_in[7];   // scalar

  long B = (long)in_sizes[0] / DP1;

  _Float16* weff = (_Float16*)d_ws;                       // 32768 B
  float*    beff = (float*)((char*)d_ws + 32768);         // 512 B

  hld_prep_weights<<<64, 256, 0, stream>>>(W_temp, b_temp, W_lt, b_lt, weff, beff);

  float* out_next = (float*)d_out;
  float* out_xc   = out_next + B * DP1;
  int blocks = (int)(B / ROWS_PER_BLOCK);                 // 262144/128 = 2048
  hld_main<<<blocks, 256, SMEM_BYTES, stream>>>(xc, xp, weff, beff,
                                                alpha, vscale, out_next, out_xc);
}
